// SARDecoder_48266842472491
// MI455X (gfx1250) — matrix operand order repaired and checked by reading, not compiled, not run on hardware
//
#include <hip/hip_runtime.h>
#include <hip/hip_bf16.h>
#include <math.h>

typedef __attribute__((ext_vector_type(16))) __bf16 v16bf;
typedef __attribute__((ext_vector_type(8)))  float  v8f;

#define RNN   512
#define ATT   512
#define FEAT  512
#define V1    111
#define STEPS 31
#define BATCH 32
#define HH    8
#define WW    32
#define HWSZ  256
#define PY    10
#define PX    34

union frag_u { uint4 u[2]; v16bf v; };

// A fragment (16x32 bf16, wave32): lane holds row (lane&15); elements 0..7 are
// k = k0+8*hi .. +7, elements 8..15 are k = k0+16+8*hi .. +7  (ISA 7.12.2)
__device__ __forceinline__ v16bf load_a_frag(const __bf16* __restrict__ row_ptr, int k0, int hi) {
  frag_u f;
  f.u[0] = *(const uint4*)(row_ptr + k0 + 8 * hi);
  f.u[1] = *(const uint4*)(row_ptr + k0 + 16 + 8 * hi);
  return f.v;
}

// B fragment from fragment-major packed weights: [tile][lane][16] bf16
__device__ __forceinline__ v16bf load_b_frag(const __bf16* __restrict__ pack, int tile, int lane) {
  frag_u f;
  const uint4* p = (const uint4*)(pack + ((size_t)tile * 32 + lane) * 16);
  f.u[0] = p[0];
  f.u[1] = p[1];
  return f.v;
}

// =======================================================================
// Weight packing: src f32 (Nlog rows x K cols, row-major "W[n][k]") ->
// fragment-major bf16: dst[((tn*K32 + kb)*32 + lane)*16 + i] = W[n][k]
// with k = kb*32 + (i&7) + 16*(i>=8) + 8*(lane>=16), n = tn*16 + (lane&15); pad n>=Nlog with 0.
// =======================================================================
__global__ void k_pack_w(const float* __restrict__ src, __bf16* __restrict__ dst,
                         int Nlog, int K, int Ntiles) {
  int idx = blockIdx.x * blockDim.x + threadIdx.x;
  int K32 = K >> 5;
  int total = Ntiles * K32 * 512;
  if (idx >= total) return;
  int i = idx & 15;
  int lane = (idx >> 4) & 31;
  int t = idx >> 9;
  int kb = t % K32;
  int tn = t / K32;
  int k = kb * 32 + (i & 7) + ((i & 8) ? 16 : 0) + ((lane & 16) ? 8 : 0);
  int n = tn * 16 + (lane & 15);
  dst[idx] = (n < Nlog) ? (__bf16)src[n * K + k] : (__bf16)0.f;
}

// Conv B pack: W_feat (ATT,FEAT,3,3) -> fragment-major with k=(ky*3+kx)*FEAT+c, n=a
__global__ void k_pack_convw(const float* __restrict__ wf, __bf16* __restrict__ dst) {
  int idx = blockIdx.x * blockDim.x + threadIdx.x;  // 32 * 144 * 512
  if (idx >= 32 * 144 * 512) return;
  int i = idx & 15;
  int lane = (idx >> 4) & 31;
  int t = idx >> 9;
  int kb = t % 144;
  int tn = t / 144;
  int k = kb * 32 + (i & 7) + ((i & 8) ? 16 : 0) + ((lane & 16) ? 8 : 0);
  int a = tn * 16 + (lane & 15);
  int tap = k >> 9;
  int c = k & 511;
  int ky = tap / 3, kx = tap - ky * 3;
  dst[idx] = (__bf16)wf[((a * FEAT + c) * 3 + ky) * 3 + kx];
}

// Padded NHWC bf16 features: featpad[b][py][px][c], halo of zeros
__global__ void k_featpad(const float* __restrict__ feat, __bf16* __restrict__ dst) {
  int idx = blockIdx.x * blockDim.x + threadIdx.x;  // 32*10*34*512
  if (idx >= BATCH * PY * PX * FEAT) return;
  int c = idx & 511;
  int r = idx >> 9;
  int px = r % PX; r /= PX;
  int py = r % PY; int b = r / PY;
  int y = py - 1, x = px - 1;
  float v = (y >= 0 && y < HH && x >= 0 && x < WW)
          ? feat[((b * FEAT + c) * HH + y) * WW + x] : 0.f;
  dst[idx] = (__bf16)v;
}

// =======================================================================
// One-time 3x3 SAME conv as implicit GEMM (M=8192 px, N=512, K=4608), WMMA.
// Each wave: one N-tile x an M-tile PAIR, sharing the B fragment between
// two WMMAs (halves weight traffic). Branch-free A gather from padded NHWC.
// =======================================================================
__global__ void k_conv_gemm(const __bf16* __restrict__ featpad, const __bf16* __restrict__ Bw,
                            const float* __restrict__ bfeat, float* __restrict__ fproj) {
  int wave = blockIdx.x * (blockDim.x >> 5) + (threadIdx.x >> 5); // 0..8191
  int lane = threadIdx.x & 31;
  int tn = wave & 31;
  int tmp = wave >> 5;           // m-tile pair 0..255
  int row_l = lane & 15;
  int hi = (lane & 16) ? 1 : 0;

  int m0 = (tmp * 2) * 16 + row_l;
  int m1 = m0 + 16;
  int b0 = m0 >> 8, hw0 = m0 & 255;
  int b1 = m1 >> 8, hw1 = m1 & 255;
  int pix0 = (b0 * PY + (hw0 >> 5)) * PX + (hw0 & 31);
  int pix1 = (b1 * PY + (hw1 >> 5)) * PX + (hw1 & 31);

  v8f acc0 = {}, acc1 = {};
#pragma unroll 2
  for (int kb = 0; kb < 144; ++kb) {
    int kk0 = kb * 32 + 8 * hi;
    int tap0 = kk0 >> 9, c0 = kk0 & 511;
    int ky0 = tap0 / 3, kx0 = tap0 - ky0 * 3;
    int kk1 = kk0 + 16;
    int tap1 = kk1 >> 9, c1 = kk1 & 511;
    int ky1 = tap1 / 3, kx1 = tap1 - ky1 * 3;
    size_t o0 = (((size_t)(ky0 * PX + kx0)) << 9) + c0;
    size_t o1 = (((size_t)(ky1 * PX + kx1)) << 9) + c1;
    frag_u fa0, fa1;
    fa0.u[0] = *(const uint4*)(featpad + ((size_t)pix0 << 9) + o0);
    fa0.u[1] = *(const uint4*)(featpad + ((size_t)pix0 << 9) + o1);
    fa1.u[0] = *(const uint4*)(featpad + ((size_t)pix1 << 9) + o0);
    fa1.u[1] = *(const uint4*)(featpad + ((size_t)pix1 << 9) + o1);
    v16bf Bf = load_b_frag(Bw, tn * 144 + kb, lane);
    acc0 = __builtin_amdgcn_wmma_f32_16x16x32_bf16(false, fa0.v, false, Bf, (short)0, acc0, false, false);
    acc1 = __builtin_amdgcn_wmma_f32_16x16x32_bf16(false, fa1.v, false, Bf, (short)0, acc1, false, false);
  }
  int a = tn * 16 + row_l;
  float bias = bfeat[a];
#pragma unroll
  for (int r = 0; r < 8; ++r) {
    int mm0 = (tmp * 2) * 16 + r + hi * 8;
    int mm1 = mm0 + 16;
    fproj[((mm0 >> 8) * ATT + a) * HWSZ + (mm0 & 255)] = acc0[r] + bias;
    fproj[((mm1 >> 8) * ATT + a) * HWSZ + (mm1 & 255)] = acc1[r] + bias;
  }
}

// =======================================================================
// Embedding gather (one-hot matmul == row gather), bf16 out
// =======================================================================
__global__ void k_embed(const float* __restrict__ Wembed, const int* __restrict__ gt,
                        int t, __bf16* __restrict__ xbf) {
  int idx = blockIdx.x * blockDim.x + threadIdx.x;
  if (idx >= BATCH * RNN) return;
  int b = idx >> 9, j = idx & 511;
  float v = 0.f;
  if (t > 0) v = Wembed[j * V1 + gt[b * STEPS + (t - 1)]];
  xbf[idx] = (__bf16)v;
}

// =======================================================================
// LSTM gate GEMM: gates(32x2048) = X@Wx + H@Wh + bi + bh   (bf16 WMMA)
// One wave per N-tile; both M-tiles per wave sharing B fragments.
// 128 waves. Grid: 16 blocks x 256 thr.
// =======================================================================
__global__ void k_lstm_gemm(const __bf16* __restrict__ x, int ldx,
                            const __bf16* __restrict__ h, int ldh,
                            const __bf16* __restrict__ Wxp, const __bf16* __restrict__ Whp,
                            const float* __restrict__ bi, const float* __restrict__ bh,
                            float* __restrict__ gates) {
  int tn = blockIdx.x * (blockDim.x >> 5) + (threadIdx.x >> 5); // 0..127
  int lane = threadIdx.x & 31;
  int row_l = lane & 15;
  int hi = (lane & 16) ? 1 : 0;
  const __bf16* xrow0 = x + (size_t)row_l * ldx;
  const __bf16* xrow1 = x + (size_t)(16 + row_l) * ldx;
  const __bf16* hrow0 = h + (size_t)row_l * ldh;
  const __bf16* hrow1 = h + (size_t)(16 + row_l) * ldh;

  v8f acc0 = {}, acc1 = {};
#pragma unroll 2
  for (int kb = 0; kb < 16; ++kb) {
    v16bf Bf = load_b_frag(Wxp, tn * 16 + kb, lane);
    v16bf A0 = load_a_frag(xrow0, kb * 32, hi);
    v16bf A1 = load_a_frag(xrow1, kb * 32, hi);
    acc0 = __builtin_amdgcn_wmma_f32_16x16x32_bf16(false, A0, false, Bf, (short)0, acc0, false, false);
    acc1 = __builtin_amdgcn_wmma_f32_16x16x32_bf16(false, A1, false, Bf, (short)0, acc1, false, false);
  }
#pragma unroll 2
  for (int kb = 0; kb < 16; ++kb) {
    v16bf Bf = load_b_frag(Whp, tn * 16 + kb, lane);
    v16bf A0 = load_a_frag(hrow0, kb * 32, hi);
    v16bf A1 = load_a_frag(hrow1, kb * 32, hi);
    acc0 = __builtin_amdgcn_wmma_f32_16x16x32_bf16(false, A0, false, Bf, (short)0, acc0, false, false);
    acc1 = __builtin_amdgcn_wmma_f32_16x16x32_bf16(false, A1, false, Bf, (short)0, acc1, false, false);
  }
  int n = tn * 16 + row_l;
  float bias = bi[n] + bh[n];
#pragma unroll
  for (int r = 0; r < 8; ++r) {
    int b = r + hi * 8;
    gates[b * 2048 + n] = acc0[r] + bias;
    gates[(b + 16) * 2048 + n] = acc1[r] + bias;
  }
}

// LSTM pointwise: update c (f32, in place) and h (bf16, strided for concat buffer)
__global__ void k_lstm_act(const float* __restrict__ gates, __bf16* __restrict__ h, int ldh,
                           float* __restrict__ c) {
  int idx = blockIdx.x * blockDim.x + threadIdx.x;
  if (idx >= BATCH * RNN) return;
  int b = idx >> 9, j = idx & 511;
  const float* g = gates + b * 2048;
  float ig = 1.f / (1.f + __expf(-g[j]));
  float fg = 1.f / (1.f + __expf(-g[RNN + j]));
  float gg = tanhf(g[2 * RNN + j]);
  float og = 1.f / (1.f + __expf(-g[3 * RNN + j]));
  float cn = fg * c[idx] + ig * gg;
  c[idx] = cn;
  h[b * ldh + j] = (__bf16)(og * tanhf(cn));
}

// Generic (32 x N) = A(32 x K bf16, stride ldA) @ Wpacked, f32 out.
// One wave per N-tile, both M-tiles share B fragments.
__global__ void k_gemm(const __bf16* __restrict__ A, int ldA, const __bf16* __restrict__ W,
                       float* __restrict__ out, int N, int K32) {
  int tn = blockIdx.x * (blockDim.x >> 5) + (threadIdx.x >> 5);
  int lane = threadIdx.x & 31;
  int row_l = lane & 15;
  int hi = (lane & 16) ? 1 : 0;
  const __bf16* arow0 = A + (size_t)row_l * ldA;
  const __bf16* arow1 = A + (size_t)(16 + row_l) * ldA;
  v8f acc0 = {}, acc1 = {};
#pragma unroll 2
  for (int kb = 0; kb < K32; ++kb) {
    v16bf Bf = load_b_frag(W, tn * K32 + kb, lane);
    v16bf A0 = load_a_frag(arow0, kb * 32, hi);
    v16bf A1 = load_a_frag(arow1, kb * 32, hi);
    acc0 = __builtin_amdgcn_wmma_f32_16x16x32_bf16(false, A0, false, Bf, (short)0, acc0, false, false);
    acc1 = __builtin_amdgcn_wmma_f32_16x16x32_bf16(false, A1, false, Bf, (short)0, acc1, false, false);
  }
  int n = tn * 16 + row_l;
#pragma unroll
  for (int r = 0; r < 8; ++r) {
    int b = r + hi * 8;
    out[b * N + n] = acc0[r];
    out[(b + 16) * N + n] = acc1[r];
  }
}

// Output projection: logits(32x111) = cat(32x1024)@Wout + b_out -> d_out[b][t][:]
// 7 N-tiles -> 7 waves (1 block), both M-tiles per wave.
__global__ void k_out_gemm(const __bf16* __restrict__ cat, const __bf16* __restrict__ Wt,
                           const float* __restrict__ bout, float* __restrict__ dout, int t) {
  int tn = threadIdx.x >> 5;
  if (tn >= 7) return;                       // wave-uniform skip
  int lane = threadIdx.x & 31;
  int row_l = lane & 15;
  int hi = (lane & 16) ? 1 : 0;
  const __bf16* arow0 = cat + (size_t)row_l * 1024;
  const __bf16* arow1 = cat + (size_t)(16 + row_l) * 1024;
  v8f acc0 = {}, acc1 = {};
#pragma unroll 2
  for (int kb = 0; kb < 32; ++kb) {
    v16bf Bf = load_b_frag(Wt, tn * 32 + kb, lane);
    v16bf A0 = load_a_frag(arow0, kb * 32, hi);
    v16bf A1 = load_a_frag(arow1, kb * 32, hi);
    acc0 = __builtin_amdgcn_wmma_f32_16x16x32_bf16(false, A0, false, Bf, (short)0, acc0, false, false);
    acc1 = __builtin_amdgcn_wmma_f32_16x16x32_bf16(false, A1, false, Bf, (short)0, acc1, false, false);
  }
  int n = tn * 16 + row_l;
  if (n < V1) {
    float bias = bout[n];
#pragma unroll
    for (int r = 0; r < 8; ++r) {
      int b = r + hi * 8;
      dout[(b * STEPS + t) * V1 + n] = acc0[r] + bias;
      dout[((b + 16) * STEPS + t) * V1 + n] = acc1[r] + bias;
    }
  }
}

// =======================================================================
// Attention: e[hw]=sum_a w_att[a]*tanh(fproj+sp); softmax; glimpse -> cat[:,512:]
// One workgroup (256 thr) per batch element.
// =======================================================================
__global__ void k_att(const float* __restrict__ fproj, const float* __restrict__ sp,
                      const float* __restrict__ watt, const float* __restrict__ features,
                      __bf16* __restrict__ cat) {
  __shared__ float sp_s[ATT];
  __shared__ float wa_s[ATT];
  __shared__ float att_s[HWSZ];
  __shared__ float red_s[HWSZ];
  int b = blockIdx.x;
  int tid = threadIdx.x;

  sp_s[tid] = sp[b * ATT + tid];
  sp_s[tid + 256] = sp[b * ATT + 256 + tid];
  wa_s[tid] = watt[tid];
  wa_s[tid + 256] = watt[tid + 256];
  __syncthreads();

  const float* fp = fproj + (size_t)b * ATT * HWSZ;
  float e = 0.f;
#pragma unroll 4
  for (int a = 0; a < ATT; ++a)
    e += wa_s[a] * tanhf(fp[a * HWSZ + tid] + sp_s[a]);

  red_s[tid] = e;
  __syncthreads();
  for (int s = 128; s > 0; s >>= 1) {
    if (tid < s) red_s[tid] = fmaxf(red_s[tid], red_s[tid + s]);
    __syncthreads();
  }
  float mx = red_s[0];
  __syncthreads();
  float ex = __expf(e - mx);
  red_s[tid] = ex;
  __syncthreads();
  for (int s = 128; s > 0; s >>= 1) {
    if (tid < s) red_s[tid] += red_s[tid + s];
    __syncthreads();
  }
  float inv = 1.f / red_s[0];
  __syncthreads();
  att_s[tid] = ex * inv;
  __syncthreads();

  for (int cc = tid; cc < FEAT; cc += 256) {
    const float* f = features + ((size_t)b * FEAT + cc) * HWSZ;
    float s = 0.f;
#pragma unroll 4
    for (int hw = 0; hw < HWSZ; ++hw) s += f[hw] * att_s[hw];
    cat[b * 1024 + 512 + cc] = (__bf16)s;   // glimpse half of concat buffer
  }
}

// =======================================================================
// Host side
// =======================================================================
extern "C" void kernel_launch(void* const* d_in, const int* in_sizes, int n_in,
                              void* d_out, int out_size, void* d_ws, size_t ws_size,
                              hipStream_t stream) {
  const float* features = (const float*)d_in[0];
  const int*   gt       = (const int*)d_in[2];
  const float* W_feat   = (const float*)d_in[3];
  const float* b_feat   = (const float*)d_in[4];
  const float* W_state  = (const float*)d_in[5];
  const float* w_att    = (const float*)d_in[6];
  const float* W_embed  = (const float*)d_in[7];
  const float* wih0     = (const float*)d_in[8];
  const float* whh0     = (const float*)d_in[9];
  const float* bih0     = (const float*)d_in[10];
  const float* bhh0     = (const float*)d_in[11];
  const float* wih1     = (const float*)d_in[12];
  const float* whh1     = (const float*)d_in[13];
  const float* bih1     = (const float*)d_in[14];
  const float* bhh1     = (const float*)d_in[15];
  const float* W_out    = (const float*)d_in[16];
  const float* b_out    = (const float*)d_in[17];
  float* out = (float*)d_out;

  char* ws = (char*)d_ws;
  size_t off = 0;
  auto take = [&](size_t bytes) { char* p = ws + off; off += (bytes + 255) & ~(size_t)255; return p; };

  __bf16* Wih0p   = (__bf16*)take(2048ull * 512 * 2);
  __bf16* Whh0p   = (__bf16*)take(2048ull * 512 * 2);
  __bf16* Wih1p   = (__bf16*)take(2048ull * 512 * 2);
  __bf16* Whh1p   = (__bf16*)take(2048ull * 512 * 2);
  __bf16* Wstatep = (__bf16*)take(512ull * 512 * 2);
  __bf16* Woutp   = (__bf16*)take(7ull * 32 * 512 * 2);
  __bf16* Wconvp  = (__bf16*)take(32ull * 144 * 512 * 2);
  __bf16* featpad = (__bf16*)take((size_t)BATCH * PY * PX * FEAT * 2);
  float*  fproj   = (float*)take(32ull * 512 * 256 * 4);
  float*  gates   = (float*)take(32ull * 2048 * 4);
  // zero-init region (contiguous): h0bf, cat(h1|glimpse), c0, c1
  __bf16* h0bf    = (__bf16*)take(32ull * 512 * 2);
  __bf16* cat     = (__bf16*)take(32ull * 1024 * 2);
  float*  c0      = (float*)take(32ull * 512 * 4);
  float*  c1      = (float*)take(32ull * 512 * 4);
  float*  sp      = (float*)take(32ull * 512 * 4);
  __bf16* xbf     = (__bf16*)take(32ull * 512 * 2);

  const int T = 256;
  // --- one-time weight prep (fragment-major packing) ---
  k_pack_w<<<(2048 * 512 + T - 1) / T, T, 0, stream>>>(wih0, Wih0p, 2048, 512, 128);
  k_pack_w<<<(2048 * 512 + T - 1) / T, T, 0, stream>>>(whh0, Whh0p, 2048, 512, 128);
  k_pack_w<<<(2048 * 512 + T - 1) / T, T, 0, stream>>>(wih1, Wih1p, 2048, 512, 128);
  k_pack_w<<<(2048 * 512 + T - 1) / T, T, 0, stream>>>(whh1, Whh1p, 2048, 512, 128);
  k_pack_w<<<(512 * 512 + T - 1) / T, T, 0, stream>>>(W_state, Wstatep, 512, 512, 32);
  k_pack_w<<<(7 * 32 * 512 + T - 1) / T, T, 0, stream>>>(W_out, Woutp, V1, 1024, 7);
  k_pack_convw<<<(32 * 144 * 512 + T - 1) / T, T, 0, stream>>>(W_feat, Wconvp);
  k_featpad<<<(BATCH * PY * PX * FEAT + T - 1) / T, T, 0, stream>>>(features, featpad);

  // zero recurrent state: h0bf + cat + c0 + c1 contiguous
  hipMemsetAsync(h0bf, 0, 32ull * 512 * 2 + 32ull * 1024 * 2 + 2 * 32ull * 512 * 4, stream);

  // --- one-time conv (implicit GEMM, WMMA) ---
  k_conv_gemm<<<1024, 256, 0, stream>>>(featpad, Wconvp, b_feat, fproj);

  // --- 31 decode steps ---
  for (int t = 0; t < STEPS; ++t) {
    k_embed<<<(32 * 512 + T - 1) / T, T, 0, stream>>>(W_embed, gt, t, xbf);
    k_lstm_gemm<<<16, 256, 0, stream>>>(xbf, 512, h0bf, 512, Wih0p, Whh0p, bih0, bhh0, gates);
    k_lstm_act<<<(32 * 512 + T - 1) / T, T, 0, stream>>>(gates, h0bf, 512, c0);
    k_lstm_gemm<<<16, 256, 0, stream>>>(h0bf, 512, cat, 1024, Wih1p, Whh1p, bih1, bhh1, gates);
    k_lstm_act<<<(32 * 512 + T - 1) / T, T, 0, stream>>>(gates, cat, 1024, c1);   // h1 -> cat[:, :512]
    k_gemm<<<4, 256, 0, stream>>>(cat, 1024, Wstatep, sp, 512, 16);
    k_att<<<32, 256, 0, stream>>>(fproj, sp, w_att, features, cat);               // glimpse -> cat[:, 512:]
    k_out_gemm<<<1, 256, 0, stream>>>(cat, Woutp, b_out, out, t);
  }
  (void)in_sizes; (void)n_in; (void)out_size; (void)ws_size;
}
